// ST_GraphFlowModel_18236431139358
// MI455X (gfx1250) — matrix operand order repaired and checked by reading, not compiled, not run on hardware
//
#include <hip/hip_runtime.h>
#include <cmath>

#define DEVINL __device__ __forceinline__

typedef __attribute__((ext_vector_type(16))) __bf16 v16bf;
typedef __attribute__((ext_vector_type(8)))  float  v8f;

namespace {
constexpr int B_ = 32, T_ = 128, J_ = 24, H_ = 256;
constexpr int BT_ = B_ * T_;        // 4096
constexpr int M_  = BT_ * J_;       // 98304
constexpr int KP_ = H_ / 2;         // 128 packed bf16-pairs per weight column

// LDS tile sizes (elements)
constexpr int SA_ELT = 64 * 32;     // bf16
constexpr int SB_ELT = 128 * 16;    // uint (packed bf16 pair)

// workspace layout (bytes)
constexpr size_t WS_H     = 0;                               // f32  [M_,H_]   (h -> h2 -> conv out, in place)
constexpr size_t WS_ABF   = WS_H     + (size_t)M_ * H_ * 4;  // bf16 [M_,H_]   (h_gcn -> h2 bf16, in place)
constexpr size_t WS_BASE  = WS_ABF   + (size_t)M_ * H_ * 2;  // f32  [BT_,H_]
constexpr size_t WS_SBF   = WS_BASE  + (size_t)BT_ * H_ * 4; // bf16 [BT_,H_]  silu(t1)
constexpr size_t WS_ZBF   = WS_SBF   + (size_t)BT_ * H_ * 2; // bf16 [BT_,H_]
constexpr size_t WS_WGCN  = WS_ZBF   + (size_t)BT_ * H_ * 2; // packed uint [H_][KP_]
constexpr size_t WS_WRAD  = WS_WGCN  + (size_t)H_ * KP_ * 4;
constexpr size_t WS_WT2   = WS_WRAD  + (size_t)H_ * KP_ * 4;
constexpr size_t WS_WCONV = WS_WT2   + (size_t)H_ * KP_ * 4; // packed uint [3][H_][KP_]
constexpr size_t WS_HRAD  = WS_WCONV + (size_t)3 * H_ * KP_ * 4; // f32 [B_,H_]
} // namespace

struct Frag { union { v16bf v; unsigned u[8]; }; };

DEVINL unsigned pack2_bf16(float a, float b) {
  union { __bf16 h[2]; unsigned u; } x;
  x.h[0] = (__bf16)a;
  x.h[1] = (__bf16)b;
  return x.u;
}

DEVINL v8f wmma_bf16(const Frag& a, const Frag& b, v8f c) {
  // (neg_a, A, neg_b, B, c_mod, C, reuse_a, reuse_b)
  return __builtin_amdgcn_wmma_f32_16x16x32_bf16(false, a.v, false, b.v,
                                                 (short)0, c, false, false);
}

// Stage one k-step of A (64x32 bf16, one b128/thread) and B (128 cols x 16
// packed uints, two b128/thread) into the given LDS buffers.
DEVINL void stage_tiles(const __bf16* __restrict__ A, const unsigned* __restrict__ WT,
                        int m0, int n0, int tap, int k0,
                        __bf16* sA, unsigned* sB,
                        int ar, int ac, int bn, int bko) {
  {
    int grow = m0 + ar;
    uint4 val = make_uint4(0u, 0u, 0u, 0u);
    if (tap >= 0) {
      int t  = (grow / J_) % T_;
      int tt = t + tap - 1;
      if (0 <= tt && tt < T_)
        val = *(const uint4*)(A + (size_t)(grow + (tap - 1) * J_) * H_ + k0 + ac);
    } else {
      val = *(const uint4*)(A + (size_t)grow * H_ + k0 + ac);
    }
    ((uint4*)sA)[threadIdx.x] = val;
  }
  {
    const uint4* s4 = (const uint4*)(WT + (size_t)(n0 + bn) * KP_ + (k0 >> 1) + bko);
    uint4* d4 = (uint4*)(sB + bn * 16 + bko);
    d4[0] = s4[0];
    d4[1] = s4[1];
  }
}

// One K=256 accumulation pass of a 64x128 C tile, double-buffered LDS pipeline.
// A  : bf16 row-major [rows][H_]
// WT : packed-transposed weights, uint [n][KP_] = (B[2kp][n], B[2kp+1][n])
// tap < 0  : plain GEMM rows m0..m0+63 of A
// tap 0..2 : conv tap — A rows shifted by (tap-1)*J_, zeroed outside [0,T_)
DEVINL void gemm_pass(const __bf16* __restrict__ A, const unsigned* __restrict__ WT,
                      int m0, int n0, int tap,
                      __bf16* sA, unsigned* sB, v8f acc[4]) {
  const int tid  = threadIdx.x;
  const int lane = tid & 31;
  const int wave = tid >> 5;
  const int half = lane >> 4;
  const int wm   = (wave & 3) * 16;   // wave's M offset in tile
  const int wn   = (wave >> 2) * 64;  // wave's N offset in tile

  // per-thread staging coordinates (constant across k-loop)
  const int ar  = tid >> 2;           // A: row 0..63
  const int ac  = (tid & 3) * 8;      // A: bf16 column offset (16B chunk)
  const int bn  = tid >> 1;           // B: local n 0..127
  const int bko = (tid & 1) * 8;      // B: packed-uint offset (two 16B chunks)

  // prologue: fill buffer 0
  stage_tiles(A, WT, m0, n0, tap, 0, sA, sB, ar, ac, bn, bko);
  __syncthreads();

  for (int k0 = 0; k0 < H_; k0 += 32) {
    const int cur = (k0 >> 5) & 1;
    __bf16*   cA  = sA + cur * SA_ELT;
    unsigned* cB  = sB + cur * SB_ELT;

    // stage next k-step into the other buffer while we compute from this one
    if (k0 + 32 < H_) {
      stage_tiles(A, WT, m0, n0, tap, k0 + 32,
                  sA + (cur ^ 1) * SA_ELT, sB + (cur ^ 1) * SB_ELT,
                  ar, ac, bn, bko);
      if (k0 + 64 < H_)  // hint the k+2 B-tile region (global_prefetch_b8)
        __builtin_prefetch(WT + (size_t)(n0 + bn) * KP_ + ((k0 + 64) >> 1), 0, 1);
    }

    // ---- A fragment (16x32): lane = row m; K contiguous per lane half (2x b128)
    Frag a;
    {
      const unsigned* rp = (const unsigned*)(cA + (size_t)(wm + (lane & 15)) * 32);
#pragma unroll
      for (int v = 0; v < 8; ++v)
        a.u[v] = rp[half * 8 + v];
    }
    // ---- 4 B fragments (32x16 each) and WMMAs (2x b128 each)
#pragma unroll
    for (int j = 0; j < 4; ++j) {
      Frag b;
      const unsigned* rp = cB + (size_t)(wn + j * 16 + (lane & 15)) * 16;
#pragma unroll
      for (int v = 0; v < 8; ++v)
        b.u[v] = rp[half * 8 + v];
      acc[j] = wmma_bf16(a, b, acc[j]);
    }
    // one barrier per iteration: orders this buffer's reads vs. its next
    // overwrite AND the other buffer's writes vs. next iteration's reads
    __syncthreads();
  }
}

// MODE 0: base = A0@W0 + A1@W1 + bias0 + bias1            (f32 out)
// MODE 1: h  += relu(A0@W0 + bias0)                       (f32 + bf16 out, in place)
// MODE 2: conv: sum of 3 shifted A0@W{0,1,2}, +bias0, BN, leaky  (f32 out)
template <int MODE>
__global__ __launch_bounds__(256) void wmma_gemm_kernel(
    const __bf16* __restrict__ A0, const unsigned* __restrict__ W0,
    const __bf16* __restrict__ A1, const unsigned* __restrict__ W1,
    const unsigned* __restrict__ W2,
    const float* __restrict__ bias0, const float* __restrict__ bias1,
    const float* __restrict__ bng, const float* __restrict__ bnb,
    const float* __restrict__ bnm, const float* __restrict__ bnv,
    const float* __restrict__ resid, float* __restrict__ outF,
    __bf16* __restrict__ outBF) {
  __shared__ __align__(16) __bf16   sA[2 * SA_ELT];
  __shared__ __align__(16) unsigned sB[2 * SB_ELT];

  const int bid  = blockIdx.x;
  const int m0   = (bid >> 1) * 64;
  const int n0   = (bid & 1) * 128;
  const int lane = threadIdx.x & 31;
  const int wave = threadIdx.x >> 5;
  const int half = lane >> 4;
  const int wm   = (wave & 3) * 16;
  const int wn   = (wave >> 2) * 64;

  v8f acc[4];
#pragma unroll
  for (int j = 0; j < 4; ++j)
#pragma unroll
    for (int i = 0; i < 8; ++i) acc[j][i] = 0.0f;

  if constexpr (MODE == 0) {
    gemm_pass(A0, W0, m0, n0, -1, sA, sB, acc);
    gemm_pass(A1, W1, m0, n0, -1, sA, sB, acc);
  } else if constexpr (MODE == 1) {
    gemm_pass(A0, W0, m0, n0, -1, sA, sB, acc);
  } else {
    gemm_pass(A0, W0, m0, n0, 0, sA, sB, acc);
    gemm_pass(A0, W1, m0, n0, 1, sA, sB, acc);
    gemm_pass(A0, W2, m0, n0, 2, sA, sB, acc);
  }

  // epilogue: C layout — VGPR i: M = i (lanes 0-15) / i+8 (lanes 16-31), N = lane&15
#pragma unroll
  for (int j = 0; j < 4; ++j) {
    int col = n0 + wn + j * 16 + (lane & 15);
#pragma unroll
    for (int i = 0; i < 8; ++i) {
      int row = m0 + wm + i + 8 * half;
      float c = acc[j][i];
      if constexpr (MODE == 0) {
        outF[(size_t)row * H_ + col] = c + bias0[col] + bias1[col];
      } else if constexpr (MODE == 1) {
        float r  = c + bias0[col];
        r        = r > 0.f ? r : 0.f;
        float hv = resid[(size_t)row * H_ + col] + r;
        outF[(size_t)row * H_ + col]  = hv;
        outBF[(size_t)row * H_ + col] = (__bf16)hv;
      } else {
        float y = c + bias0[col];
        float s = bng[col] * rsqrtf(bnv[col] + 1e-5f);
        y       = (y - bnm[col]) * s + bnb[col];
        y       = y >= 0.f ? y : 0.2f * y;
        outF[(size_t)row * H_ + col] = y;
      }
    }
  }
}

// weights -> packed-transposed bf16-pair layout WT[n][kp] = (B[2kp][n], B[2kp+1][n])
// conv: B_tap[k][n] = w_conv[n][k][tap]  (w_conv is [out][in][3] row-major)
// gemm weights stored [in][out]: B[k][n] = w[k*H_+n]
__global__ void prep_weights_kernel(const float* __restrict__ w_gcn,
                                    const float* __restrict__ w_radar,
                                    const float* __restrict__ w_t2,
                                    const float* __restrict__ w_conv,
                                    unsigned* wgcnT, unsigned* wradT,
                                    unsigned* wt2T, unsigned* wconvT) {
  int stride = gridDim.x * blockDim.x;
  const int PER = H_ * KP_;  // 32768 packed uints per matrix
  for (int i = blockIdx.x * blockDim.x + threadIdx.x; i < 3 * PER; i += stride) {
    int tap = i / PER;
    int r   = i % PER;
    int n   = r >> 7;        // 0..255 (out channel)
    int kp  = r & 127;       // packed K pair
    wconvT[i] = pack2_bf16(w_conv[(size_t)n * (H_ * 3) + (2 * kp) * 3 + tap],
                           w_conv[(size_t)n * (H_ * 3) + (2 * kp + 1) * 3 + tap]);
    if (i < PER) {
      wgcnT[i] = pack2_bf16(w_gcn[(size_t)(2 * kp) * H_ + n],
                            w_gcn[(size_t)(2 * kp + 1) * H_ + n]);
      wradT[i] = pack2_bf16(w_radar[(size_t)(2 * kp) * H_ + n],
                            w_radar[(size_t)(2 * kp + 1) * H_ + n]);
      wt2T[i]  = pack2_bf16(w_t2[(size_t)(2 * kp) * H_ + n],
                            w_t2[(size_t)(2 * kp + 1) * H_ + n]);
    }
  }
}

// s = silu(tau*w_t1 + b_t1) -> bf16 ; z -> bf16
__global__ void prep_act_kernel(const float* __restrict__ tau,
                                const float* __restrict__ w_t1,
                                const float* __restrict__ b_t1,
                                const float* __restrict__ z,
                                __bf16* sbf, __bf16* zbf) {
  int idx = blockIdx.x * blockDim.x + threadIdx.x;  // exactly BT_*H_
  int r = idx >> 8, c = idx & 255;
  float u = tau[r] * w_t1[c] + b_t1[c];
  float s = u / (1.f + __expf(-u));
  sbf[idx] = (__bf16)s;
  zbf[idx] = (__bf16)z[idx];
}

__global__ void hradar_kernel(const float* __restrict__ h_n,
                              const float* __restrict__ w_radar,
                              const float* __restrict__ b_radar,
                              float* __restrict__ hrad) {
  int b = blockIdx.x, c = threadIdx.x;
  float acc = b_radar[c];
  for (int k = 0; k < H_; ++k) acc += h_n[b * H_ + k] * w_radar[k * H_ + c];
  hrad[b * H_ + c] = acc;
}

// h = x@w_joint + b_joint + base[bt] + hrad[b]
__global__ void assemble_h_kernel(const float* __restrict__ xt,
                                  const float* __restrict__ w_joint,
                                  const float* __restrict__ b_joint,
                                  const float* __restrict__ base,
                                  const float* __restrict__ hrad,
                                  float* __restrict__ h) {
  int row = blockIdx.x;  // (b*T+t)*J + j
  int c   = threadIdx.x;
  int bt  = row / J_;
  int b   = bt / T_;
  const float* x3 = xt + (size_t)row * 3;
  float v = x3[0] * w_joint[c] + x3[1] * w_joint[H_ + c] + x3[2] * w_joint[2 * H_ + c]
          + b_joint[c] + base[(size_t)bt * H_ + c] + hrad[(size_t)b * H_ + c];
  h[(size_t)row * H_ + c] = v;
}

// h_gcn[bt][j][:] = sum_k adj[j][k] * h[bt][k][:]  -> bf16
__global__ void gcn_mix_kernel(const float* __restrict__ adj,
                               const float* __restrict__ h,
                               __bf16* __restrict__ abf) {
  __shared__ float sh[J_][H_];
  __shared__ float sadj[J_ * J_];
  int bt = blockIdx.x;
  int c  = threadIdx.x;
#pragma unroll
  for (int r = 0; r < J_; ++r)
    sh[r][c] = h[((size_t)bt * J_ + r) * H_ + c];
  for (int i = c; i < J_ * J_; i += 256) sadj[i] = adj[i];
  __syncthreads();
  for (int j = 0; j < J_; ++j) {
    float acc = 0.f;
#pragma unroll
    for (int k = 0; k < J_; ++k) acc += sadj[j * J_ + k] * sh[k][c];
    abf[((size_t)bt * J_ + j) * H_ + c] = (__bf16)acc;
  }
}

// v = h @ w_out + b_out ; wave-per-row, shuffle reduction (wave32)
__global__ void out_proj_kernel(const float* __restrict__ h,
                                const float* __restrict__ w_out,
                                const float* __restrict__ b_out,
                                float* __restrict__ out) {
  int lane = threadIdx.x & 31;
  int wave = threadIdx.x >> 5;
  int row  = blockIdx.x * 8 + wave;
  const float* hr = h + (size_t)row * H_;
  float a0 = 0.f, a1 = 0.f, a2 = 0.f;
#pragma unroll
  for (int it = 0; it < 8; ++it) {
    int c   = lane + it * 32;
    float v = hr[c];
    a0 += v * w_out[c * 3 + 0];
    a1 += v * w_out[c * 3 + 1];
    a2 += v * w_out[c * 3 + 2];
  }
#pragma unroll
  for (int off = 16; off > 0; off >>= 1) {
    a0 += __shfl_xor(a0, off, 32);
    a1 += __shfl_xor(a1, off, 32);
    a2 += __shfl_xor(a2, off, 32);
  }
  if (lane == 0) {
    out[(size_t)row * 3 + 0] = a0 + b_out[0];
    out[(size_t)row * 3 + 1] = a1 + b_out[1];
    out[(size_t)row * 3 + 2] = a2 + b_out[2];
  }
}

extern "C" void kernel_launch(void* const* d_in, const int* in_sizes, int n_in,
                              void* d_out, int out_size, void* d_ws, size_t ws_size,
                              hipStream_t stream) {
  const float* xt      = (const float*)d_in[0];
  const float* tau     = (const float*)d_in[1];
  const float* z       = (const float*)d_in[2];
  const float* h_n     = (const float*)d_in[3];
  const float* adj     = (const float*)d_in[4];
  const float* w_joint = (const float*)d_in[5];
  const float* b_joint = (const float*)d_in[6];
  const float* w_t1    = (const float*)d_in[7];
  const float* b_t1    = (const float*)d_in[8];
  const float* w_t2    = (const float*)d_in[9];
  const float* b_t2    = (const float*)d_in[10];
  const float* w_radar = (const float*)d_in[11];
  const float* b_radar = (const float*)d_in[12];
  const float* w_gcn   = (const float*)d_in[13];
  const float* b_gcn   = (const float*)d_in[14];
  const float* w_conv  = (const float*)d_in[15];
  const float* b_conv  = (const float*)d_in[16];
  const float* bng     = (const float*)d_in[17];
  const float* bnb     = (const float*)d_in[18];
  const float* bnm     = (const float*)d_in[19];
  const float* bnv     = (const float*)d_in[20];
  const float* w_out   = (const float*)d_in[21];
  const float* b_out   = (const float*)d_in[22];

  char* ws = (char*)d_ws;
  float*    hF    = (float*)(ws + WS_H);
  __bf16*   aBF   = (__bf16*)(ws + WS_ABF);
  float*    base  = (float*)(ws + WS_BASE);
  __bf16*   sBF   = (__bf16*)(ws + WS_SBF);
  __bf16*   zBF   = (__bf16*)(ws + WS_ZBF);
  unsigned* wgcnT = (unsigned*)(ws + WS_WGCN);
  unsigned* wradT = (unsigned*)(ws + WS_WRAD);
  unsigned* wt2T  = (unsigned*)(ws + WS_WT2);
  unsigned* wcvT  = (unsigned*)(ws + WS_WCONV);
  float*    hrad  = (float*)(ws + WS_HRAD);

  prep_weights_kernel<<<384, 256, 0, stream>>>(w_gcn, w_radar, w_t2, w_conv,
                                               wgcnT, wradT, wt2T, wcvT);
  prep_act_kernel<<<BT_ * H_ / 256, 256, 0, stream>>>(tau, w_t1, b_t1, z, sBF, zBF);
  hradar_kernel<<<B_, H_, 0, stream>>>(h_n, w_radar, b_radar, hrad);

  // base = silu(t1)@w_t2 + z@w_radar + b_t2 + b_radar
  wmma_gemm_kernel<0><<<(BT_ / 64) * 2, 256, 0, stream>>>(
      sBF, wt2T, zBF, wradT, nullptr, b_t2, b_radar,
      nullptr, nullptr, nullptr, nullptr, nullptr, base, nullptr);

  assemble_h_kernel<<<M_, H_, 0, stream>>>(xt, w_joint, b_joint, base, hrad, hF);
  gcn_mix_kernel<<<BT_, H_, 0, stream>>>(adj, hF, aBF);

  // h += relu(h_gcn @ w_gcn + b_gcn); dual f32/bf16 write-back (in place)
  wmma_gemm_kernel<1><<<(M_ / 64) * 2, 256, 0, stream>>>(
      aBF, wgcnT, nullptr, nullptr, nullptr, b_gcn, nullptr,
      nullptr, nullptr, nullptr, nullptr, hF, hF, aBF);

  // temporal conv (3 shifted GEMMs) + bias + BN + leaky-relu
  wmma_gemm_kernel<2><<<(M_ / 64) * 2, 256, 0, stream>>>(
      aBF, wcvT, nullptr, wcvT + (size_t)H_ * KP_, wcvT + (size_t)2 * H_ * KP_,
      b_conv, nullptr, bng, bnb, bnm, bnv, nullptr, hF, nullptr);

  out_proj_kernel<<<M_ / 8, 256, 0, stream>>>(hF, w_out, b_out, (float*)d_out);
}
